// MyPVT2_77567109365876
// MI455X (gfx1250) — matrix operand order repaired and checked by reading, not compiled, not run on hardware
//
#include <hip/hip_runtime.h>
#include <math.h>
#include <stdint.h>

// ---------------------------------------------------------------------------
// Problem constants (MyPVT2)
// ---------------------------------------------------------------------------
constexpr int Bc   = 32;
constexpr int NGc  = 49;
constexpr int NAc  = 3136;
constexpr int Cc   = 256;
constexpr int DOc  = 512;
constexpr int NHc  = 8;
constexpr int Sc   = 784;
constexpr int Hc   = 56;
constexpr int Wc   = 56;
constexpr int HIDc = 2048;
constexpr int NSRC = NGc + NAc;          // 3185
constexpr float SCALEc = 0.17677669529663687f;  // (C/NH)^-0.5 = 32^-0.5
constexpr float EPSc = 1e-5f;

typedef __bf16 bf16;
typedef __attribute__((ext_vector_type(16))) __bf16 v16bf;
typedef __attribute__((ext_vector_type(8)))  float  v8f;

#define DEVI __device__ __forceinline__

DEVI v8f vzero8() { v8f z = {0.f,0.f,0.f,0.f,0.f,0.f,0.f,0.f}; return z; }

DEVI v8f wmma_bf16(v16bf a, v16bf b, v8f c) {
  return __builtin_amdgcn_wmma_f32_16x16x32_bf16(false, a, false, b, (short)0, c, false, false);
}

// CDNA5 async global -> LDS copy (VGPR-bypassing DMA path, ASYNCcnt-tracked).
// lds_addr: 32-bit LDS byte address (low 32 bits of flat LDS pointer per ISA
// flat-aperture rules); gaddr: 64-bit global address. 16B granularity.
DEVI void async_b128(uint32_t lds_addr, const void* gaddr) {
  asm volatile("global_load_async_to_lds_b128 %0, %1, off"
               :: "v"(lds_addr), "v"(gaddr) : "memory");
}
DEVI void wait_async0() {
  asm volatile("s_wait_asynccnt 0x0" ::: "memory");
}
DEVI uint32_t lds_addr_of(const void* p) { return (uint32_t)(uintptr_t)p; }

// XOR lane shuffle via ds_swizzle (wave32, group-of-32 mode: and=0x1f, xor=XM)
template<int XM>
DEVI float xshf(float v) {
  int i = __builtin_amdgcn_ds_swizzle(__float_as_int(v), (XM << 10) | 0x1f);
  return __int_as_float(i);
}
DEVI float redsum16(float v) { v += xshf<1>(v); v += xshf<2>(v); v += xshf<4>(v); v += xshf<8>(v); return v; }
DEVI float redmax16(float v) {
  v = fmaxf(v, xshf<1>(v)); v = fmaxf(v, xshf<2>(v));
  v = fmaxf(v, xshf<4>(v)); v = fmaxf(v, xshf<8>(v)); return v;
}
DEVI float redsum32(float v) { v = redsum16(v); v += xshf<16>(v); return v; }
DEVI float redmax32(float v) { v = redmax16(v); v = fmaxf(v, xshf<16>(v)); return v; }

// ---------------------------------------------------------------------------
// Generic tiled WMMA GEMM:  out[M,N] = A[M,K](bf16) @ W[K,N](bf16) (+bias)(+act)(+add)
// Block = 256 threads = 8 waves, 64x64 tile, K-step 32. Double-buffered LDS
// tiles filled with global_load_async_to_lds_b128 (loads for step k+1 overlap
// the WMMAs of step k). Each wave: 16x32 slice, 2 accumulators.
// Requires N % 64 == 0 and K % 32 == 0 (true for all uses here).
// ---------------------------------------------------------------------------
template<bool BIAS, int ACT, bool ADD, bool OUT32, bool OUT16>
__global__ __launch_bounds__(256)
void gemm_bf16(const bf16* __restrict__ A, const bf16* __restrict__ W,
               const float* __restrict__ bias, const float* __restrict__ addsrc,
               float* __restrict__ out32, bf16* __restrict__ out16,
               int M, int N, int K)
{
  __shared__ bf16 As[2][64][40];   // row stride 80B (16B aligned)
  __shared__ bf16 Bs[2][32][72];   // row stride 144B (16B aligned)
  const int tid  = threadIdx.x;
  const int lane = tid & 31;
  const int wv   = tid >> 5;
  const int half = lane >> 4, l16 = lane & 15;
  const int wm   = (wv & 3) * 16;
  const int wn   = (wv >> 2) * 32;
  const int m0   = blockIdx.y * 64, n0 = blockIdx.x * 64;

  v8f acc0 = vzero8(), acc1 = vzero8();

  const int ar = tid >> 2, ac = (tid & 3) * 8;   // A staging: 64x32, 16B per thread
  const int br = tid >> 3, bc = (tid & 7) * 8;   // B staging: 32x64, 16B per thread
  int gm = m0 + ar; if (gm >= M) gm = M - 1;      // clamp: rows >= M feed unstored C rows

  const bf16* aPtr = A + (size_t)gm * K + ac;
  const bf16* bPtr = W + (size_t)br * N + n0 + bc;

  const int nk = K >> 5;
  // Prologue: fill buffer 0
  async_b128(lds_addr_of(&As[0][ar][ac]), aPtr);
  async_b128(lds_addr_of(&Bs[0][br][bc]), bPtr);

  for (int kk = 0; kk < nk; kk++) {
    const int buf = kk & 1;
    wait_async0();
    __syncthreads();
    if (kk + 1 < nk) {           // prefetch next K-step into other buffer
      int k1 = (kk + 1) << 5;
      async_b128(lds_addr_of(&As[buf ^ 1][ar][ac]), aPtr + k1);
      async_b128(lds_addr_of(&Bs[buf ^ 1][br][bc]), bPtr + (size_t)k1 * N);
    }

    v16bf af;
    {
      int row = wm + l16, kb = half * 16;
      #pragma unroll
      for (int i = 0; i < 8; i++) af[i]     = As[buf][row][kb + i];
      #pragma unroll
      for (int i = 0; i < 8; i++) af[8 + i] = As[buf][row][kb + 8 + i];
    }
    v16bf b0, b1;
    {
      int kb = half * 16;
      #pragma unroll
      for (int i = 0; i < 16; i++) {
        b0[i] = Bs[buf][kb + i][wn + l16];
        b1[i] = Bs[buf][kb + i][wn + 16 + l16];
      }
    }
    acc0 = wmma_bf16(af, b0, acc0);
    acc1 = wmma_bf16(af, b1, acc1);
  }

  #pragma unroll
  for (int t = 0; t < 2; t++) {
    v8f acc = t ? acc1 : acc0;
    int n = n0 + wn + t * 16 + l16;
    #pragma unroll
    for (int i = 0; i < 8; i++) {
      int m = m0 + wm + half * 8 + i;
      if (m >= M) continue;
      float v = acc[i];
      if (BIAS) v += bias[n];
      if (ACT == 1) v = 0.5f * v * (1.0f + erff(v * 0.70710678118654752f));  // exact GELU
      if (ADD) v += addsrc[(size_t)m * N + n];
      if (OUT32) out32[(size_t)m * N + n] = v;
      if (OUT16) out16[(size_t)m * N + n] = (bf16)v;
    }
  }
}

// ---------------------------------------------------------------------------
// Flash attention (WMMA, online softmax). Per block: 64 query rows, one head,
// one batch. 4 waves x 16 q-rows. K/V tiles of 32 keys double-buffered in LDS
// via async global->LDS loads. Q/K/V/O are bf16 [T,512], head = 64-col slice.
// ---------------------------------------------------------------------------
__global__ __launch_bounds__(128)
void attn_kernel(const bf16* __restrict__ Qg, const bf16* __restrict__ Kg,
                 const bf16* __restrict__ Vg, bf16* __restrict__ Og,
                 int Tq, int Tk)
{
  __shared__ bf16 Ks[2][32][72];
  __shared__ bf16 Vs[2][32][72];
  __shared__ bf16 Ps[4][16][40];
  const int tid  = threadIdx.x;
  const int lane = tid & 31, wv = tid >> 5;
  const int half = lane >> 4, l16 = lane & 15;
  const int hoff = blockIdx.y * 64;
  const int z    = blockIdx.z;
  const bf16* Q = Qg + (size_t)z * Tq * DOc;
  const bf16* K = Kg + (size_t)z * Tk * DOc;
  const bf16* V = Vg + (size_t)z * Tk * DOc;
  bf16*       O = Og + (size_t)z * Tq * DOc;
  const int q0 = blockIdx.x * 64 + wv * 16;

  // Async staging of one 32-key K/V block (32B per thread per matrix).
  const int skey = tid >> 2, scol = (tid & 3) * 16;
  auto issueKV = [&](int kb, int buf) {
    int gk = kb * 32 + skey; if (gk >= Tk) gk = Tk - 1;   // clamped keys get masked
    const bf16* kp = K + (size_t)gk * DOc + hoff + scol;
    const bf16* vp = V + (size_t)gk * DOc + hoff + scol;
    async_b128(lds_addr_of(&Ks[buf][skey][scol]), kp);
    async_b128(lds_addr_of(&Ks[buf][skey][scol + 8]), kp + 8);
    async_b128(lds_addr_of(&Vs[buf][skey][scol]), vp);
    async_b128(lds_addr_of(&Vs[buf][skey][scol + 8]), vp + 8);
  };

  // Preload Q fragments (d = 0..31 and 32..63)
  v16bf qa[2];
  {
    int qrow = q0 + l16;
    bool ok = qrow < Tq;
    const bf16* qp = Q + (size_t)(ok ? qrow : 0) * DOc + hoff;
    #pragma unroll
    for (int d = 0; d < 2; d++) {
      int kb = d * 32 + half * 16;
      #pragma unroll
      for (int i = 0; i < 8; i++) qa[d][i]     = ok ? qp[kb + i]      : (bf16)0.f;
      #pragma unroll
      for (int i = 0; i < 8; i++) qa[d][8 + i] = ok ? qp[kb + 8 + i]  : (bf16)0.f;
    }
  }

  v8f o0 = vzero8(), o1 = vzero8(), o2 = vzero8(), o3 = vzero8();
  float mrun[8], lrun[8];
  #pragma unroll
  for (int i = 0; i < 8; i++) { mrun[i] = -1e30f; lrun[i] = 0.f; }

  const int nkb = (Tk + 31) >> 5;
  issueKV(0, 0);
  for (int kb0 = 0; kb0 < nkb; kb0++) {
    const int buf = kb0 & 1;
    wait_async0();
    __syncthreads();
    if (kb0 + 1 < nkb) issueKV(kb0 + 1, buf ^ 1);

    // S = Q K^T  (two 16-key tiles, two 32-d k-steps)
    v8f s0 = vzero8(), s1 = vzero8();
    #pragma unroll
    for (int d = 0; d < 2; d++) {
      v16bf bk0, bk1;
      #pragma unroll
      for (int i = 0; i < 16; i++) {
        bk0[i] = Ks[buf][l16][d * 32 + half * 16 + i];
        bk1[i] = Ks[buf][16 + l16][d * 32 + half * 16 + i];
      }
      s0 = wmma_bf16(qa[d], bk0, s0);
      s1 = wmma_bf16(qa[d], bk1, s1);
    }

    // Online softmax per row (rows live in 16-lane halves)
    int k0g = kb0 * 32 + l16;
    bool v0ok = k0g < Tk, v1ok = (k0g + 16) < Tk;
    #pragma unroll
    for (int i = 0; i < 8; i++) {
      float a = v0ok ? s0[i] * SCALEc : -1e30f;
      float c = v1ok ? s1[i] * SCALEc : -1e30f;
      float rm = redmax16(fmaxf(a, c));
      float mn = fmaxf(mrun[i], rm);
      float p0 = __expf(a - mn);
      float p1 = __expf(c - mn);
      float rs = redsum16(p0 + p1);
      float corr = __expf(mrun[i] - mn);
      lrun[i] = lrun[i] * corr + rs;
      mrun[i] = mn;
      o0[i] *= corr; o1[i] *= corr; o2[i] *= corr; o3[i] *= corr;
      Ps[wv][half * 8 + i][l16]      = (bf16)p0;  // transpose C-layout -> A-layout
      Ps[wv][half * 8 + i][16 + l16] = (bf16)p1;
    }

    // P @ V  (P as 16x32 A-fragment, V block as four 32x16 B-fragments)
    v16bf pa;
    {
      int kb = half * 16;
      #pragma unroll
      for (int i = 0; i < 8; i++) pa[i]     = Ps[wv][l16][kb + i];
      #pragma unroll
      for (int i = 0; i < 8; i++) pa[8 + i] = Ps[wv][l16][kb + 8 + i];
    }
    {
      v16bf bv;
      #pragma unroll
      for (int i = 0; i < 16; i++) bv[i] = Vs[buf][half * 16 + i][0 + l16];
      o0 = wmma_bf16(pa, bv, o0);
      #pragma unroll
      for (int i = 0; i < 16; i++) bv[i] = Vs[buf][half * 16 + i][16 + l16];
      o1 = wmma_bf16(pa, bv, o1);
      #pragma unroll
      for (int i = 0; i < 16; i++) bv[i] = Vs[buf][half * 16 + i][32 + l16];
      o2 = wmma_bf16(pa, bv, o2);
      #pragma unroll
      for (int i = 0; i < 16; i++) bv[i] = Vs[buf][half * 16 + i][48 + l16];
      o3 = wmma_bf16(pa, bv, o3);
    }
  }

  #pragma unroll
  for (int i = 0; i < 8; i++) {
    int row = q0 + half * 8 + i;
    if (row < Tq) {
      float inv = 1.0f / lrun[i];
      bf16* op = O + (size_t)row * DOc + hoff;
      op[0  + l16] = (bf16)(o0[i] * inv);
      op[16 + l16] = (bf16)(o1[i] * inv);
      op[32 + l16] = (bf16)(o2[i] * inv);
      op[48 + l16] = (bf16)(o3[i] * inv);
    }
  }
}

// ---------------------------------------------------------------------------
// Confidence logits: LN(x_ada; dn) @ conf_w + conf_b. One wave per token.
// ---------------------------------------------------------------------------
__global__ __launch_bounds__(256)
void conf_logits(const float* __restrict__ x, const float* __restrict__ g,
                 const float* __restrict__ bt, const float* __restrict__ w,
                 const float* __restrict__ cb, float* __restrict__ out)
{
  int lane = threadIdx.x & 31, wv = threadIdx.x >> 5;
  int tok = blockIdx.x * 8 + wv;
  if (tok >= Bc * NAc) return;
  const float* xp = x + (size_t)tok * Cc;
  float v[8]; float s = 0.f;
  #pragma unroll
  for (int i = 0; i < 8; i++) { v[i] = xp[i * 32 + lane]; s += v[i]; }
  s = redsum32(s);
  float mean = s * (1.0f / Cc);
  float var = 0.f;
  #pragma unroll
  for (int i = 0; i < 8; i++) { float d = v[i] - mean; var += d * d; }
  var = redsum32(var);
  float rstd = rsqrtf(var * (1.0f / Cc) + EPSc);
  float dot = 0.f;
  #pragma unroll
  for (int i = 0; i < 8; i++) {
    int c = i * 32 + lane;
    dot += ((v[i] - mean) * rstd * g[c] + bt[c]) * w[c];
  }
  dot = redsum32(dot);
  if (lane == 0) out[tok] = dot + cb[0];
}

// Softmax over NA per batch, in place, then scale by NA.
__global__ __launch_bounds__(256)
void conf_softmax(float* __restrict__ conf)
{
  __shared__ float red[8];
  __shared__ float stat[2];
  float* p = conf + (size_t)blockIdx.x * NAc;
  int tid = threadIdx.x, lane = tid & 31, wv = tid >> 5;
  float mx = -1e30f;
  for (int i = tid; i < NAc; i += 256) mx = fmaxf(mx, p[i]);
  mx = redmax32(mx);
  if (lane == 0) red[wv] = mx;
  __syncthreads();
  if (tid == 0) { float m = red[0]; for (int i = 1; i < 8; i++) m = fmaxf(m, red[i]); stat[0] = m; }
  __syncthreads();
  float m = stat[0];
  float sm = 0.f;
  for (int i = tid; i < NAc; i += 256) sm += __expf(p[i] - m);
  sm = redsum32(sm);
  if (lane == 0) red[wv] = sm;
  __syncthreads();
  if (tid == 0) { float t = 0.f; for (int i = 0; i < 8; i++) t += red[i]; stat[1] = (float)NAc / t; }
  __syncthreads();
  float sc = stat[1];
  for (int i = tid; i < NAc; i += 256) p[i] = __expf(p[i] - m) * sc;
}

// Top-k via full in-LDS bitonic sort (4096 padded, descending). One block/batch.
__global__ __launch_bounds__(1024)
void topk_kernel(const float* __restrict__ conf, int* __restrict__ idx)
{
  __shared__ float key[4096];
  __shared__ int   val[4096];
  int tid = threadIdx.x;
  const float* p = conf + (size_t)blockIdx.x * NAc;
  for (int i = tid; i < 4096; i += 1024) {
    key[i] = (i < NAc) ? p[i] : -1e30f;
    val[i] = i;
  }
  __syncthreads();
  for (int k = 2; k <= 4096; k <<= 1) {
    for (int j = k >> 1; j > 0; j >>= 1) {
      for (int ii = tid; ii < 2048; ii += 1024) {
        int i  = ((ii & ~(j - 1)) << 1) | (ii & (j - 1));
        int p2 = i | j;
        bool dirDesc = ((i & k) == 0);
        float a = key[i], c = key[p2];
        bool sw = dirDesc ? (a < c) : (a > c);
        if (sw) {
          key[i] = c; key[p2] = a;
          int t = val[i]; val[i] = val[p2]; val[p2] = t;
        }
      }
      __syncthreads();
    }
  }
  for (int s = tid; s < Sc; s += 1024) idx[blockIdx.x * Sc + s] = val[s];
}

// Generic LayerNorm f32 -> bf16, optional row gather + per-token scale.
__global__ __launch_bounds__(256)
void ln_kernel(const float* __restrict__ x, int Tin, int Tout, int D,
               const int* __restrict__ gidx, const float* __restrict__ scale,
               const float* __restrict__ g, const float* __restrict__ bt,
               bf16* __restrict__ out, int outStride, int nrows)
{
  int lane = threadIdx.x & 31, wv = threadIdx.x >> 5;
  int r = blockIdx.x * 8 + wv;
  if (r >= nrows) return;
  int b = r / Tout, t = r - b * Tout;
  int j = gidx ? gidx[b * Tout + t] : t;
  const float* xp = x + ((size_t)b * Tin + j) * D;
  int E = D >> 5;
  float v[16];
  float s = 0.f;
  for (int e = 0; e < E; e++) { v[e] = xp[e * 32 + lane]; s += v[e]; }
  s = redsum32(s);
  float mean = s / (float)D;
  float var = 0.f;
  for (int e = 0; e < E; e++) { float d = v[e] - mean; var += d * d; }
  var = redsum32(var);
  float rstd = rsqrtf(var / (float)D + EPSc);
  float sc = scale ? scale[(size_t)b * Tin + j] : 1.0f;
  bf16* op = out + ((size_t)b * outStride + t) * D;
  for (int e = 0; e < E; e++) {
    int c = e * 32 + lane;
    op[c] = (bf16)(((v[e] - mean) * rstd * g[c] + bt[c]) * sc);
  }
}

__global__ void f32_to_bf16(const float* __restrict__ in, bf16* __restrict__ out, size_t n) {
  size_t i = (size_t)blockIdx.x * blockDim.x + threadIdx.x;
  if (i < n) out[i] = (bf16)in[i];
}

__global__ void gather_rows_bf16(const float* __restrict__ in, const int* __restrict__ idx,
                                 bf16* __restrict__ out, int Tin, int Tout, int D) {
  int r = blockIdx.x;
  int b = r / Tout, t = r - b * Tout;
  int j = idx[b * Tout + t];
  const float* ip = in + ((size_t)b * Tin + j) * D;
  bf16* op = out + (size_t)r * D;
  for (int c = threadIdx.x; c < D; c += blockDim.x) op[c] = (bf16)ip[c];
}

__global__ void gather_loc(const float* __restrict__ loc_ada, const int* __restrict__ idx,
                           float* __restrict__ out) {
  int i = blockIdx.x * blockDim.x + threadIdx.x;
  if (i >= Bc * Sc) return;
  int b = i / Sc;
  int j = idx[i];
  out[(size_t)i * 2 + 0] = loc_ada[((size_t)b * NAc + j) * 2 + 0];
  out[(size_t)i * 2 + 1] = loc_ada[((size_t)b * NAc + j) * 2 + 1];
}

// Bilinear grid_sample of pos_embed [H*W, DO], add into outputs.
__global__ __launch_bounds__(128)
void pos_kernel(const float* __restrict__ pe, const float* __restrict__ loc_grid,
                const float* __restrict__ loc_down,
                float* __restrict__ out_g, float* __restrict__ out_a)
{
  int r = blockIdx.x;                       // 0 .. B*(NG+S)-1
  int b = r / (NGc + Sc), t = r - b * (NGc + Sc);
  const float* lp; float* dst;
  if (t < NGc) { lp = loc_grid + ((size_t)b * NGc + t) * 2; dst = out_g + ((size_t)b * NGc + t) * DOc; }
  else { int s = t - NGc; lp = loc_down + ((size_t)b * Sc + s) * 2; dst = out_a + ((size_t)b * Sc + s) * DOc; }
  float gx = lp[0] * 2.f - 1.f, gy = lp[1] * 2.f - 1.f;
  float ix = ((gx + 1.f) * Wc - 1.f) * 0.5f;
  float iy = ((gy + 1.f) * Hc - 1.f) * 0.5f;
  float x0 = floorf(ix), y0 = floorf(iy);
  float wx1 = ix - x0, wx0 = 1.f - wx1;
  float wy1 = iy - y0, wy0 = 1.f - wy1;
  int xi0 = (int)x0, yi0 = (int)y0, xi1 = xi0 + 1, yi1 = yi0 + 1;
  auto valid = [](int xi, int yi) { return (xi >= 0 && xi < Wc && yi >= 0 && yi < Hc) ? 1.f : 0.f; };
  float w00 = wx0 * wy0 * valid(xi0, yi0);
  float w10 = wx1 * wy0 * valid(xi1, yi0);
  float w01 = wx0 * wy1 * valid(xi0, yi1);
  float w11 = wx1 * wy1 * valid(xi1, yi1);
  int cx0 = min(max(xi0, 0), Wc - 1), cx1 = min(max(xi1, 0), Wc - 1);
  int cy0 = min(max(yi0, 0), Hc - 1), cy1 = min(max(yi1, 0), Hc - 1);
  const float* p00 = pe + ((size_t)cy0 * Wc + cx0) * DOc;
  const float* p10 = pe + ((size_t)cy0 * Wc + cx1) * DOc;
  const float* p01 = pe + ((size_t)cy1 * Wc + cx0) * DOc;
  const float* p11 = pe + ((size_t)cy1 * Wc + cx1) * DOc;
  for (int d = threadIdx.x; d < DOc; d += blockDim.x)
    dst[d] += p00[d] * w00 + p10[d] * w10 + p01[d] * w01 + p11[d] * w11;
}

// ---------------------------------------------------------------------------
// Host orchestration
// ---------------------------------------------------------------------------
extern "C" void kernel_launch(void* const* d_in, const int* in_sizes, int n_in,
                              void* d_out, int out_size, void* d_ws, size_t ws_size,
                              hipStream_t stream) {
  const float* x_grid    = (const float*)d_in[0];
  const float* x_ada     = (const float*)d_in[1];
  const float* loc_grid  = (const float*)d_in[2];
  const float* loc_ada   = (const float*)d_in[3];
  const float* pos_embed = (const float*)d_in[4];
  const float* dn_g      = (const float*)d_in[5];
  const float* dn_b      = (const float*)d_in[6];
  const float* conf_w    = (const float*)d_in[7];
  const float* conf_b    = (const float*)d_in[8];
  const float* n1_g      = (const float*)d_in[9];
  const float* n1_b      = (const float*)d_in[10];
  const float* qw        = (const float*)d_in[11];
  const float* kw        = (const float*)d_in[12];
  const float* vw        = (const float*)d_in[13];
  const float* k2w       = (const float*)d_in[14];
  const float* v2w       = (const float*)d_in[15];
  const float* pw        = (const float*)d_in[16];
  const float* pb        = (const float*)d_in[17];
  const float* fcw       = (const float*)d_in[18];
  const float* fcb       = (const float*)d_in[19];
  const float* n2_g      = (const float*)d_in[20];
  const float* n2_b      = (const float*)d_in[21];
  const float* m1w       = (const float*)d_in[22];
  const float* m1b       = (const float*)d_in[23];
  const float* m2w       = (const float*)d_in[24];
  const float* m2b       = (const float*)d_in[25];

  float* og   = (float*)d_out;                                  // [B*NG, DO]
  float* oa   = og + (size_t)Bc * NGc * DOc;                    // [B*S, DO]
  float* oloc = oa + (size_t)Bc * Sc * DOc;                     // [B*S, 2]

  char* ws = (char*)d_ws;
  size_t off = 0;
  auto take = [&](size_t bytes) { size_t o = (off + 255) & ~(size_t)255; off = o + bytes; return o; };

  size_t o_conf  = take((size_t)Bc * NAc * 4);
  size_t o_idx   = take((size_t)Bc * Sc * 4);
  size_t o_qwb   = take((size_t)Cc * DOc * 2);
  size_t o_kwb   = take((size_t)Cc * DOc * 2);
  size_t o_vwb   = take((size_t)Cc * DOc * 2);
  size_t o_k2wb  = take((size_t)DOc * DOc * 2);
  size_t o_v2wb  = take((size_t)DOc * DOc * 2);
  size_t o_pwb   = take((size_t)DOc * DOc * 2);
  size_t o_fcwb  = take((size_t)Cc * DOc * 2);
  size_t o_m1wb  = take((size_t)DOc * HIDc * 2);
  size_t o_m2wb  = take((size_t)HIDc * DOc * 2);
  size_t o_xgn   = take((size_t)Bc * NGc * Cc * 2);
  size_t o_src   = take((size_t)Bc * NSRC * Cc * 2);
  size_t o_xdn   = take((size_t)Bc * Sc * Cc * 2);
  size_t o_xgraw = take((size_t)Bc * NGc * Cc * 2);
  size_t o_xdraw = take((size_t)Bc * Sc * Cc * 2);
  size_t o_qgrid = take((size_t)Bc * NGc * DOc * 2);
  size_t o_qdown = take((size_t)Bc * Sc * DOc * 2);
  size_t o_big   = take(2ull * Bc * NSRC * DOc * 2);            // 208.7 MB, reused
  size_t o_ogrid = take((size_t)Bc * NGc * DOc * 2);
  size_t o_ygrid = take((size_t)Bc * NGc * DOc * 4);
  size_t o_ygb   = take((size_t)Bc * NGc * DOc * 2);
  size_t o_k2b   = take((size_t)Bc * NGc * DOc * 2);
  size_t o_v2b   = take((size_t)Bc * NGc * DOc * 2);
  size_t o_xg32  = take((size_t)Bc * NGc * DOc * 4);
  size_t o_tg    = take((size_t)Bc * NGc * DOc * 2);
  size_t o_hg    = take((size_t)Bc * NGc * HIDc * 2);
  // Aliases (phase-disjoint lifetimes):
  size_t o_ksrc  = o_big;                                        // phase A
  size_t o_vsrc  = o_big + (size_t)Bc * NSRC * DOc * 2;
  size_t o_oada  = o_big;                                        // phase B
  size_t o_yada  = ((o_oada + (size_t)Bc * Sc * DOc * 2 + 255) & ~(size_t)255);
  size_t o_ha    = ((o_yada + (size_t)Bc * Sc * DOc * 4 + 255) & ~(size_t)255);
  size_t o_xa32  = o_src;                                        // src dead after projections
  size_t o_ta    = o_qdown;                                      // q_down dead after attn2

  float* conf   = (float*)(ws + o_conf);
  int*   idxp   = (int*)(ws + o_idx);
  bf16*  qwb    = (bf16*)(ws + o_qwb);
  bf16*  kwb    = (bf16*)(ws + o_kwb);
  bf16*  vwb    = (bf16*)(ws + o_vwb);
  bf16*  k2wb   = (bf16*)(ws + o_k2wb);
  bf16*  v2wb   = (bf16*)(ws + o_v2wb);
  bf16*  pwb    = (bf16*)(ws + o_pwb);
  bf16*  fcwb   = (bf16*)(ws + o_fcwb);
  bf16*  m1wb   = (bf16*)(ws + o_m1wb);
  bf16*  m2wb   = (bf16*)(ws + o_m2wb);
  bf16*  xgn    = (bf16*)(ws + o_xgn);
  bf16*  srcb   = (bf16*)(ws + o_src);
  bf16*  xdn    = (bf16*)(ws + o_xdn);
  bf16*  xgraw  = (bf16*)(ws + o_xgraw);
  bf16*  xdraw  = (bf16*)(ws + o_xdraw);
  bf16*  qgrid  = (bf16*)(ws + o_qgrid);
  bf16*  qdown  = (bf16*)(ws + o_qdown);
  bf16*  ksrc   = (bf16*)(ws + o_ksrc);
  bf16*  vsrc   = (bf16*)(ws + o_vsrc);
  bf16*  ogrid  = (bf16*)(ws + o_ogrid);
  float* ygrid  = (float*)(ws + o_ygrid);
  bf16*  ygb    = (bf16*)(ws + o_ygb);
  bf16*  k2b    = (bf16*)(ws + o_k2b);
  bf16*  v2b    = (bf16*)(ws + o_v2b);
  bf16*  oada   = (bf16*)(ws + o_oada);
  float* yada   = (float*)(ws + o_yada);
  float* xg32   = (float*)(ws + o_xg32);
  float* xa32   = (float*)(ws + o_xa32);
  bf16*  tg     = (bf16*)(ws + o_tg);
  bf16*  ta     = (bf16*)(ws + o_ta);
  bf16*  hg     = (bf16*)(ws + o_hg);
  bf16*  ha     = (bf16*)(ws + o_ha);

  auto conv = [&](const float* s, bf16* d, size_t n) {
    f32_to_bf16<<<(int)((n + 255) / 256), 256, 0, stream>>>(s, d, n);
  };

  // 1. Weight conversion f32 -> bf16
  conv(qw,  qwb,  (size_t)Cc * DOc);
  conv(kw,  kwb,  (size_t)Cc * DOc);
  conv(vw,  vwb,  (size_t)Cc * DOc);
  conv(k2w, k2wb, (size_t)DOc * DOc);
  conv(v2w, v2wb, (size_t)DOc * DOc);
  conv(pw,  pwb,  (size_t)DOc * DOc);
  conv(fcw, fcwb, (size_t)Cc * DOc);
  conv(m1w, m1wb, (size_t)DOc * HIDc);
  conv(m2w, m2wb, (size_t)HIDc * DOc);

  // 2-4. Confidence -> softmax -> top-k indices
  conf_logits<<<(Bc * NAc) / 8, 256, 0, stream>>>(x_ada, dn_g, dn_b, conf_w, conf_b, conf);
  conf_softmax<<<Bc, 256, 0, stream>>>(conf);
  topk_kernel<<<Bc, 1024, 0, stream>>>(conf, idxp);

  // 5. LayerNorms (n1) -> bf16 activations
  ln_kernel<<<(Bc * NGc + 7) / 8, 256, 0, stream>>>(x_grid, NGc, NGc, Cc, nullptr, nullptr,
      n1_g, n1_b, xgn, NGc, Bc * NGc);
  ln_kernel<<<(Bc * NGc + 7) / 8, 256, 0, stream>>>(x_grid, NGc, NGc, Cc, nullptr, nullptr,
      n1_g, n1_b, srcb, NSRC, Bc * NGc);                              // src[:, :49]
  ln_kernel<<<(Bc * NAc + 7) / 8, 256, 0, stream>>>(x_ada, NAc, NAc, Cc, nullptr, conf,
      n1_g, n1_b, srcb + (size_t)NGc * Cc, NSRC, Bc * NAc);           // src[:, 49:] * conf
  ln_kernel<<<(Bc * Sc + 7) / 8, 256, 0, stream>>>(x_ada, NAc, Sc, Cc, idxp, nullptr,
      n1_g, n1_b, xdn, Sc, Bc * Sc);                                  // LN of selected tokens

  // 6. Raw copies / gathers
  conv(x_grid, xgraw, (size_t)Bc * NGc * Cc);
  gather_rows_bf16<<<Bc * Sc, 256, 0, stream>>>(x_ada, idxp, xdraw, NAc, Sc, Cc);
  gather_loc<<<(Bc * Sc + 255) / 256, 256, 0, stream>>>(loc_ada, idxp, oloc);

  // 7. Projections (WMMA GEMMs)
  gemm_bf16<false,0,false,false,true><<<dim3(DOc/64, (Bc*NGc+63)/64), 256, 0, stream>>>(
      xgn, qwb, nullptr, nullptr, nullptr, qgrid, Bc*NGc, DOc, Cc);
  gemm_bf16<false,0,false,false,true><<<dim3(DOc/64, (Bc*NSRC+63)/64), 256, 0, stream>>>(
      srcb, kwb, nullptr, nullptr, nullptr, ksrc, Bc*NSRC, DOc, Cc);
  gemm_bf16<false,0,false,false,true><<<dim3(DOc/64, (Bc*NSRC+63)/64), 256, 0, stream>>>(
      srcb, vwb, nullptr, nullptr, nullptr, vsrc, Bc*NSRC, DOc, Cc);
  gemm_bf16<false,0,false,false,true><<<dim3(DOc/64, (Bc*Sc+63)/64), 256, 0, stream>>>(
      xdn, qwb, nullptr, nullptr, nullptr, qdown, Bc*Sc, DOc, Cc);

  // 8. Attention 1 (grid queries over full src)
  attn_kernel<<<dim3(1, NHc, Bc), 128, 0, stream>>>(qgrid, ksrc, vsrc, ogrid, NGc, NSRC);

  // 9. Output projection -> y_grid (f32 for residual/K2V2 source) + bf16
  gemm_bf16<true,0,false,true,true><<<dim3(DOc/64, (Bc*NGc+63)/64), 256, 0, stream>>>(
      ogrid, pwb, pb, nullptr, ygrid, ygb, Bc*NGc, DOc, DOc);

  // 10. K2/V2 from y_grid
  gemm_bf16<false,0,false,false,true><<<dim3(DOc/64, (Bc*NGc+63)/64), 256, 0, stream>>>(
      ygb, k2wb, nullptr, nullptr, nullptr, k2b, Bc*NGc, DOc, DOc);
  gemm_bf16<false,0,false,false,true><<<dim3(DOc/64, (Bc*NGc+63)/64), 256, 0, stream>>>(
      ygb, v2wb, nullptr, nullptr, nullptr, v2b, Bc*NGc, DOc, DOc);

  // 11. Attention 2 (selected-token queries over grid output)
  attn_kernel<<<dim3((Sc+63)/64, NHc, Bc), 128, 0, stream>>>(qdown, k2b, v2b, oada, Sc, NGc);

  // 12. Output projection -> y_ada
  gemm_bf16<true,0,false,true,false><<<dim3(DOc/64, (Bc*Sc+63)/64), 256, 0, stream>>>(
      oada, pwb, pb, nullptr, yada, nullptr, Bc*Sc, DOc, DOc);

  // 13. fc residual: x @ fcw + fcb + y
  gemm_bf16<true,0,true,true,false><<<dim3(DOc/64, (Bc*NGc+63)/64), 256, 0, stream>>>(
      xgraw, fcwb, fcb, ygrid, xg32, nullptr, Bc*NGc, DOc, Cc);
  gemm_bf16<true,0,true,true,false><<<dim3(DOc/64, (Bc*Sc+63)/64), 256, 0, stream>>>(
      xdraw, fcwb, fcb, yada, xa32, nullptr, Bc*Sc, DOc, Cc);

  // 14. LN (n2)
  ln_kernel<<<(Bc * NGc + 7) / 8, 256, 0, stream>>>(xg32, NGc, NGc, DOc, nullptr, nullptr,
      n2_g, n2_b, tg, NGc, Bc * NGc);
  ln_kernel<<<(Bc * Sc + 7) / 8, 256, 0, stream>>>(xa32, Sc, Sc, DOc, nullptr, nullptr,
      n2_g, n2_b, ta, Sc, Bc * Sc);

  // 15. MLP up + GELU
  gemm_bf16<true,1,false,false,true><<<dim3(HIDc/64, (Bc*NGc+63)/64), 256, 0, stream>>>(
      tg, m1wb, m1b, nullptr, nullptr, hg, Bc*NGc, HIDc, DOc);
  gemm_bf16<true,1,false,false,true><<<dim3(HIDc/64, (Bc*Sc+63)/64), 256, 0, stream>>>(
      ta, m1wb, m1b, nullptr, nullptr, ha, Bc*Sc, HIDc, DOc);

  // 16. MLP down + residual -> final outputs into d_out
  gemm_bf16<true,0,true,true,false><<<dim3(DOc/64, (Bc*NGc+63)/64), 256, 0, stream>>>(
      hg, m2wb, m2b, xg32, og, nullptr, Bc*NGc, DOc, HIDc);
  gemm_bf16<true,0,true,true,false><<<dim3(DOc/64, (Bc*Sc+63)/64), 256, 0, stream>>>(
      ha, m2wb, m2b, xa32, oa, nullptr, Bc*Sc, DOc, HIDc);

  // 17. Positional embedding (bilinear grid_sample) added in place
  pos_kernel<<<Bc * (NGc + Sc), 128, 0, stream>>>(pos_embed, loc_grid, oloc, og, oa);
}
